// GIN_87823491268919
// MI455X (gfx1250) — compile-verified, matrix operand order checked by reading, not run on hardware
//
#include <hip/hip_runtime.h>

#define N_NODES 20000
#define DEG_K   16
#define Q_EDGES 16384
#define IN_DIM  128
#define HID     512

typedef __attribute__((ext_vector_type(16))) __bf16 v16bf;
typedef __attribute__((ext_vector_type(8)))  float  v8f;

#define F_RELU 1
#define F_BN   2
#define F_LN   4

union FragU { v16bf v; unsigned u[8]; };

// Load a 16x32 bf16 fragment (A layout per CDNA5 ISA 7.12.2) from a row-major
// bf16 matrix: lane holds row (lane&15); K halves split by lane group.
// Lowered by the compiler to two global_load_b128.
__device__ inline v16bf load_frag_bf16(const __bf16* __restrict__ base,
                                       int row0, int k0, int ld, int lane) {
  int r  = row0 + (lane & 15);
  int kb = k0 + ((lane >> 4) << 3);
  const __bf16* p = base + (size_t)r * ld + kb;
  FragU f;
  f.u[0] = *(const unsigned*)(p + 0);
  f.u[1] = *(const unsigned*)(p + 2);
  f.u[2] = *(const unsigned*)(p + 4);
  f.u[3] = *(const unsigned*)(p + 6);
  f.u[4] = *(const unsigned*)(p + 16);
  f.u[5] = *(const unsigned*)(p + 18);
  f.u[6] = *(const unsigned*)(p + 20);
  f.u[7] = *(const unsigned*)(p + 22);
  return f.v;
}

// ---------------------------------------------------------------------------
// Fused WMMA GEMM for Nout == 512 exactly:
//   out = epilogue(A[M,K] @ Wt^T + bias)
//   Wt: pre-packed transposed weights Wt[n][k] bf16 row-major, n = 0..511.
//   Block = 16 rows x 512 cols; 8 waves x 4 col tiles each. Branch-free k-loop.
//   Epilogues: RELU / relu-then-BN / LN-then-relu, then y*scale + residual.
// ---------------------------------------------------------------------------
__global__ __launch_bounds__(256) void gemm_wmma_bf16(
    const __bf16* __restrict__ A, const __bf16* __restrict__ Wt,
    const float* __restrict__ bias,
    int K, int flags,
    const float* __restrict__ gamma, const float* __restrict__ betav,
    const float* __restrict__ rmean, const float* __restrict__ rvar,
    const float* __restrict__ res, const float* __restrict__ scale_ptr,
    __bf16* __restrict__ out_bf, float* __restrict__ out_f32)
{
  __shared__ float s_sum[16];
  __shared__ float s_sq[16];
  const int tid  = threadIdx.x;
  const int lane = tid & 31;
  const int wave = tid >> 5;
  const int row0 = blockIdx.x << 4;

  if (tid < 16) { s_sum[tid] = 0.f; s_sq[tid] = 0.f; }
  __syncthreads();

  const int nt0 = wave << 2;          // this wave's first 16-col tile (0..28)
  const int nb0 = nt0 << 4;           // first output column of this wave

  v8f acc0 = (v8f){0,0,0,0,0,0,0,0};
  v8f acc1 = acc0, acc2 = acc0, acc3 = acc0;

  #pragma unroll 2
  for (int kt = 0; kt < K; kt += 32) {
    v16bf a  = load_frag_bf16(A,  row0,     kt, K, lane);
    v16bf b0 = load_frag_bf16(Wt, nb0,      kt, K, lane);
    v16bf b1 = load_frag_bf16(Wt, nb0 + 16, kt, K, lane);
    v16bf b2 = load_frag_bf16(Wt, nb0 + 32, kt, K, lane);
    v16bf b3 = load_frag_bf16(Wt, nb0 + 48, kt, K, lane);
    acc0 = __builtin_amdgcn_wmma_f32_16x16x32_bf16(false, a, false, b0, (short)0, acc0, false, false);
    acc1 = __builtin_amdgcn_wmma_f32_16x16x32_bf16(false, a, false, b1, (short)0, acc1, false, false);
    acc2 = __builtin_amdgcn_wmma_f32_16x16x32_bf16(false, a, false, b2, (short)0, acc2, false, false);
    acc3 = __builtin_amdgcn_wmma_f32_16x16x32_bf16(false, a, false, b3, (short)0, acc3, false, false);
  }

  // ---- epilogue: D layout = vgpr i, lane l -> m = i + 8*(l>=16), n = l&15
  const int mbase = (lane >> 4) << 3;
  const int nlow  = lane & 15;
  float y[4][8];
  #pragma unroll
  for (int t = 0; t < 4; ++t) {
    int n = nb0 + (t << 4) + nlow;
    float bv = bias[n];
    v8f a = (t == 0) ? acc0 : (t == 1) ? acc1 : (t == 2) ? acc2 : acc3;
    #pragma unroll
    for (int i = 0; i < 8; ++i) y[t][i] = a[i] + bv;
  }

  if (flags & F_BN) {
    #pragma unroll
    for (int t = 0; t < 4; ++t) {
      int n = nb0 + (t << 4) + nlow;
      float sc = rsqrtf(rvar[n] + 1e-5f) * gamma[n];
      float mn = rmean[n], bb = betav[n];
      #pragma unroll
      for (int i = 0; i < 8; ++i) {
        float v = y[t][i];
        if (flags & F_RELU) v = fmaxf(v, 0.f);   // reference: relu BEFORE bn
        y[t][i] = (v - mn) * sc + bb;
      }
    }
  } else if (flags & F_LN) {
    #pragma unroll
    for (int i = 0; i < 8; ++i) {
      float ps = 0.f, pq = 0.f;
      #pragma unroll
      for (int t = 0; t < 4; ++t) { float v = y[t][i]; ps += v; pq += v * v; }
      atomicAdd(&s_sum[mbase + i], ps);
      atomicAdd(&s_sq[mbase + i], pq);
    }
    __syncthreads();
    float mu[8], rs[8];
    #pragma unroll
    for (int i = 0; i < 8; ++i) {
      float su = s_sum[mbase + i], sq = s_sq[mbase + i];
      float m_ = su * (1.f / (float)HID);
      float var = sq * (1.f / (float)HID) - m_ * m_;
      mu[i] = m_;
      rs[i] = rsqrtf(var + 1e-5f);
    }
    #pragma unroll
    for (int t = 0; t < 4; ++t) {
      int n = nb0 + (t << 4) + nlow;
      float gg = gamma[n], bb = betav[n];
      #pragma unroll
      for (int i = 0; i < 8; ++i) {
        float v = (y[t][i] - mu[i]) * rs[i] * gg + bb;
        if (flags & F_RELU) v = fmaxf(v, 0.f);   // reference: ln THEN relu
        y[t][i] = v;
      }
    }
  } else if (flags & F_RELU) {
    #pragma unroll
    for (int t = 0; t < 4; ++t)
      #pragma unroll
      for (int i = 0; i < 8; ++i) y[t][i] = fmaxf(y[t][i], 0.f);
  }

  float scl = (scale_ptr != nullptr) ? scale_ptr[0] : 1.f;
  #pragma unroll
  for (int t = 0; t < 4; ++t) {
    int n = nb0 + (t << 4) + nlow;
    #pragma unroll
    for (int i = 0; i < 8; ++i) {
      int m = row0 + mbase + i;
      float v = y[t][i] * scl;
      if (res != nullptr) v += res[(size_t)m * HID + n];
      if (out_bf  != nullptr) out_bf[(size_t)m * HID + n] = (__bf16)v;
      if (out_f32 != nullptr) out_f32[(size_t)m * HID + n] = v;
    }
  }
}

// ---------------------------------------------------------------------------
// Head: out[Q,7] = A[Q,512] @ Wf3 + bf3.  Wt padded to 16 cols (zeros).
// Each wave owns its own 16-row tile -> no inter-wave coupling, no branches
// around WMMA. grid = Q/128, block = 256 (8 waves).
// ---------------------------------------------------------------------------
__global__ __launch_bounds__(256) void head_wmma_bf16(
    const __bf16* __restrict__ A, const __bf16* __restrict__ Wt,
    const float* __restrict__ bias, float* __restrict__ out)
{
  const int lane = threadIdx.x & 31;
  const int wave = threadIdx.x >> 5;
  const int row0 = (blockIdx.x * 8 + wave) << 4;

  v8f acc = (v8f){0,0,0,0,0,0,0,0};
  #pragma unroll 2
  for (int kt = 0; kt < HID; kt += 32) {
    v16bf a = load_frag_bf16(A,  row0, kt, HID, lane);
    v16bf b = load_frag_bf16(Wt, 0,    kt, HID, lane);
    acc = __builtin_amdgcn_wmma_f32_16x16x32_bf16(false, a, false, b, (short)0, acc, false, false);
  }
  const int n     = lane & 15;
  const int mbase = (lane >> 4) << 3;
  if (n < 7) {
    float bv = bias[n];
    #pragma unroll
    for (int i = 0; i < 8; ++i)
      out[(size_t)(row0 + mbase + i) * 7 + n] = acc[i] + bv;
  }
}

// ---- weight pack: W[k][Nout] f32 -> Wt[n][k] bf16 (n padded to NoutPad) ----
__global__ void pack_weights(const float* __restrict__ W, __bf16* __restrict__ Wt,
                             int K, int Nout, int NoutPad) {
  int idx = blockIdx.x * 256 + threadIdx.x;
  if (idx >= NoutPad * K) return;
  int n = idx / K, k = idx % K;
  float v = (n < Nout) ? W[(size_t)k * Nout + n] : 0.f;
  Wt[idx] = (__bf16)v;
}

// ---- GIN aggregation 1: z0 = (1+eps)*x + sum_d x[nbr[i,d]]  (f32 -> bf16) ----
__global__ __launch_bounds__(256) void agg1_kernel(
    const float* __restrict__ x, const int* __restrict__ nbr,
    const float* __restrict__ eps, __bf16* __restrict__ z0) {
  int row = blockIdx.x * 2 + (threadIdx.x >> 7);
  int col = threadIdx.x & 127;
  float s = (1.f + eps[0]) * x[(size_t)row * IN_DIM + col];
  #pragma unroll
  for (int d = 0; d < DEG_K; ++d) {
    int nb = nbr[row * DEG_K + d];
    s += x[(size_t)nb * IN_DIM + col];
  }
  z0[(size_t)row * IN_DIM + col] = (__bf16)s;
}

// ---- GIN aggregation 2: z2 = (1+eps)*h + sum_d h[nbr[i,d]]  (bf16 -> bf16) ----
__global__ __launch_bounds__(256) void agg2_kernel(
    const __bf16* __restrict__ h, const int* __restrict__ nbr,
    const float* __restrict__ eps, __bf16* __restrict__ z2) {
  int row = blockIdx.x;
  int c   = threadIdx.x * 2;
  float e = 1.f + eps[0];
  float s0 = e * (float)h[(size_t)row * HID + c];
  float s1 = e * (float)h[(size_t)row * HID + c + 1];
  #pragma unroll
  for (int d = 0; d < DEG_K; ++d) {
    int nb = nbr[row * DEG_K + d];
    s0 += (float)h[(size_t)nb * HID + c];
    s1 += (float)h[(size_t)nb * HID + c + 1];
  }
  z2[(size_t)row * HID + c]     = (__bf16)s0;
  z2[(size_t)row * HID + c + 1] = (__bf16)s1;
}

// ---- edge gather: xij_in[q] = h3[i]*h3[j] (f32 -> bf16) ----
__global__ __launch_bounds__(256) void xij_gather_kernel(
    const float* __restrict__ h3, const int* __restrict__ pos_edge,
    __bf16* __restrict__ out) {
  int q = blockIdx.x;
  int c = threadIdx.x * 2;
  int i = pos_edge[q];
  int j = pos_edge[Q_EDGES + q];
  out[(size_t)q * HID + c]     = (__bf16)(h3[(size_t)i * HID + c]     * h3[(size_t)j * HID + c]);
  out[(size_t)q * HID + c + 1] = (__bf16)(h3[(size_t)i * HID + c + 1] * h3[(size_t)j * HID + c + 1]);
}

// ---- common-neighbor masked sum: xcn[q] = sum_d mask[d] * h4[ni[q,d]] ----
__global__ __launch_bounds__(256) void xcn_kernel(
    const __bf16* __restrict__ h4, const int* __restrict__ nbr,
    const int* __restrict__ pos_edge, __bf16* __restrict__ xcn) {
  __shared__ int   s_ni[DEG_K];
  __shared__ int   s_nj[DEG_K];
  __shared__ float s_mask[DEG_K];
  int q = blockIdx.x, tid = threadIdx.x;
  if (tid < DEG_K) {
    s_ni[tid] = nbr[pos_edge[q] * DEG_K + tid];
    s_nj[tid] = nbr[pos_edge[Q_EDGES + q] * DEG_K + tid];
  }
  __syncthreads();
  if (tid < DEG_K) {
    float m = 0.f;
    #pragma unroll
    for (int dp = 0; dp < DEG_K; ++dp) if (s_ni[tid] == s_nj[dp]) m = 1.f;
    s_mask[tid] = m;
  }
  __syncthreads();
  int c = tid * 2;
  float s0 = 0.f, s1 = 0.f;
  #pragma unroll
  for (int d = 0; d < DEG_K; ++d) {
    if (s_mask[d] != 0.f) {
      int r = s_ni[d];
      s0 += (float)h4[(size_t)r * HID + c];
      s1 += (float)h4[(size_t)r * HID + c + 1];
    }
  }
  xcn[(size_t)q * HID + c]     = (__bf16)s0;
  xcn[(size_t)q * HID + c + 1] = (__bf16)s1;
}

// ---------------------------------------------------------------------------
extern "C" void kernel_launch(void* const* d_in, const int* in_sizes, int n_in,
                              void* d_out, int out_size, void* d_ws, size_t ws_size,
                              hipStream_t stream) {
  auto P = [&](int i) -> const float* { return (const float*)d_in[i]; };
  const float* x        = (const float*)d_in[0];
  const int*   nbr      = (const int*)d_in[2];
  const int*   pos_edge = (const int*)d_in[3];
  const float* eps1     = P(4);
  const float* eps2     = P(5);

  // bump allocator over workspace
  char* ws = (char*)d_ws;
  size_t off = 0;
  auto alloc = [&](size_t bytes) -> void* {
    off = (off + 255) & ~(size_t)255;
    void* p = ws + off;
    off += bytes;
    return p;
  };

  // --- pack all weights to bf16 transposed layout -------------------------
  struct WDesc { int idx; int K; int Nout; int NoutPad; };
  const WDesc wd[14] = {
    {6, IN_DIM, HID, HID},  // W1a
    {8,  HID, HID, HID},    // W1b
    {14, HID, HID, HID},    // W2a
    {20, HID, HID, HID},    // Wl1
    {22, HID, HID, HID},    // Wx1
    {24, HID, HID, HID},    // Wx2
    {36, HID, HID, HID},    // Wij1
    {40, HID, HID, HID},    // Wij2
    {28, HID, HID, HID},    // Wc1
    {30, HID, HID, HID},    // Wc2
    {34, HID, HID, HID},    // Wc3
    {42, HID, HID, HID},    // Wf1
    {46, HID, HID, HID},    // Wf2
    {50, HID, 7, 16},       // Wf3 (padded to 16 cols)
  };
  __bf16* wt[14];
  for (int i = 0; i < 14; ++i) {
    size_t cnt = (size_t)wd[i].NoutPad * wd[i].K;
    wt[i] = (__bf16*)alloc(cnt * sizeof(__bf16));
    pack_weights<<<(int)((cnt + 255) / 256), 256, 0, stream>>>(
        P(wd[i].idx), wt[i], wd[i].K, wd[i].Nout, wd[i].NoutPad);
  }

  // --- activation buffers -------------------------------------------------
  __bf16* z0   = (__bf16*)alloc((size_t)N_NODES * IN_DIM * 2);
  __bf16* bufA = (__bf16*)alloc((size_t)N_NODES * HID * 2);
  __bf16* bufB = (__bf16*)alloc((size_t)N_NODES * HID * 2);
  __bf16* bufC = (__bf16*)alloc((size_t)N_NODES * HID * 2);
  float*  h3f  = (float*) alloc((size_t)N_NODES * HID * 4);
  __bf16* qA   = (__bf16*)alloc((size_t)Q_EDGES * HID * 2);
  __bf16* qB   = (__bf16*)alloc((size_t)Q_EDGES * HID * 2);
  __bf16* qC   = (__bf16*)alloc((size_t)Q_EDGES * HID * 2);
  float*  xijf = (float*) alloc((size_t)Q_EDGES * HID * 4);

  auto gemm = [&](const __bf16* A, int wi, const float* bias, int M,
                  int flags, const float* g, const float* b,
                  const float* mv, const float* vv,
                  const float* res, const float* scale,
                  __bf16* obf, float* of32) {
    gemm_wmma_bf16<<<M / 16, 256, 0, stream>>>(
        A, wt[wi], bias, wd[wi].K, flags,
        g, b, mv, vv, res, scale, obf, of32);
  };

  // --- node-side pipeline -------------------------------------------------
  agg1_kernel<<<N_NODES / 2, 256, 0, stream>>>(x, nbr, eps1, z0);
  // t1 = relu(z0 @ W1a + b1a)
  gemm(z0, 0, P(7), N_NODES, F_RELU, 0, 0, 0, 0, 0, 0, bufA, 0);
  // h1 = bn1(relu(t1 @ W1b + b1b))
  gemm(bufA, 1, P(9), N_NODES, F_RELU | F_BN, P(10), P(11), P(12), P(13),
       0, 0, bufB, 0);
  // z2 = (1+eps2)*h1 + agg(h1)
  agg2_kernel<<<N_NODES, 256, 0, stream>>>(bufB, nbr, eps2, bufA);
  // h2 = bn2(relu(z2 @ W2a + b2a))
  gemm(bufA, 2, P(15), N_NODES, F_RELU | F_BN, P(16), P(17), P(18), P(19),
       0, 0, bufC, 0);
  // h3 = h2 @ Wl1 + bl1   (bf16 + f32 copies)
  gemm(bufC, 3, P(21), N_NODES, 0, 0, 0, 0, 0, 0, 0, bufA, h3f);
  // a1 = relu(h3 @ Wx1 + bx1)
  gemm(bufA, 4, P(23), N_NODES, F_RELU, 0, 0, 0, 0, 0, 0, bufB, 0);
  // h4 = h3 + relu(ln(a1 @ Wx2 + bx2))
  gemm(bufB, 5, P(25), N_NODES, F_RELU | F_LN, P(26), P(27), 0, 0,
       h3f, 0, bufC, 0);

  // --- edge-side pipeline -------------------------------------------------
  xij_gather_kernel<<<Q_EDGES, 256, 0, stream>>>(h3f, pos_edge, qA);
  xcn_kernel<<<Q_EDGES, 256, 0, stream>>>(bufC, nbr, pos_edge, qB);
  // u1 = relu(ln(xij_in @ Wij1 + bij1))
  gemm(qA, 6, P(37), Q_EDGES, F_RELU | F_LN, P(38), P(39), 0, 0,
       0, 0, qC, 0);
  // xij = u1 @ Wij2 + bij2  (f32)
  gemm(qC, 7, P(41), Q_EDGES, 0, 0, 0, 0, 0, 0, 0, nullptr, xijf);
  // c1 = relu(xcn @ Wc1 + bc1)
  gemm(qB, 8, P(29), Q_EDGES, F_RELU, 0, 0, 0, 0, 0, 0, qA, 0);
  // c2 = relu(ln(c1 @ Wc2 + bc2))
  gemm(qA, 9, P(31), Q_EDGES, F_RELU | F_LN, P(32), P(33), 0, 0,
       0, 0, qC, 0);
  // z = (c2 @ Wc3 + bc3) * beta + xij
  gemm(qC, 10, P(35), Q_EDGES, 0, 0, 0, 0, 0, xijf, P(52), qA, 0);
  // f1 = relu(ln(z @ Wf1 + bf1))
  gemm(qA, 11, P(43), Q_EDGES, F_RELU | F_LN, P(44), P(45), 0, 0,
       0, 0, qB, 0);
  // f2 = relu(ln(f1 @ Wf2 + bf2))
  gemm(qB, 12, P(47), Q_EDGES, F_RELU | F_LN, P(48), P(49), 0, 0,
       0, 0, qA, 0);
  // out = f2 @ Wf3 + bf3  (f32, 7 cols)
  head_wmma_bf16<<<Q_EDGES / 128, 256, 0, stream>>>(qA, wt[13], P(51), (float*)d_out);
}
